// BalancedRLIFLayer_89618787598521
// MI455X (gfx1250) — matrix operand order repaired and checked by reading, not compiled, not run on hardware
//
#include <hip/hip_runtime.h>
#include <hip/hip_bf16.h>

// ---------------------------------------------------------------------------
// BalancedRLIF layer for MI455X (gfx1250, wave32, WMMA).
//   Phase 1: Wx = x @ W.T   -> v_wmma_f32_16x16x32_bf16; x tile staged to LDS
//            as bf16 once per block (8 waves share it), W bf16 from L2.
//   Phase 2: sequential scan, 2 workgroups (16 batch rows each), per-step
//            [16,512]x[512,512] recurrent GEMM via WMMA, o in LDS (bf16),
//            membrane v carried in C/D-layout registers.
// ---------------------------------------------------------------------------

typedef __attribute__((ext_vector_type(16))) __bf16 v16bf;
typedef __attribute__((ext_vector_type(8)))  __bf16 v8bf;
typedef __attribute__((ext_vector_type(8)))  float  v8f;

union AFrag { v16bf v; v8bf h[2]; };           // assemble A from two 16B loads

#define Bsz 32
#define Tsz 2000
#define Hsz 512
#define Isz 512

#define DECAY 0.8f          // 1 - h*lambda_v
#define HSTEP 0.01f
#define NSCALE 0.001f       // sigma_v * sqrt(h)
#define INVH 100.0f         // 1/h (exact in f32 rounding)

static __device__ __forceinline__ unsigned short bf16_bits(float f) {
  unsigned u = __builtin_bit_cast(unsigned, f);
  u += 0x7FFFu + ((u >> 16) & 1u);            // round-to-nearest-even
  return (unsigned short)(u >> 16);
}

static __device__ __forceinline__ __bf16 to_bf16(float f) {
  unsigned short s = bf16_bits(f);
  return __builtin_bit_cast(__bf16, s);
}

static __device__ __forceinline__ v8f zero8() {
  v8f z = {0.f, 0.f, 0.f, 0.f, 0.f, 0.f, 0.f, 0.f};
  return z;
}

static __device__ __forceinline__ v8f wmma_bf16(v16bf a, v16bf b, v8f c) {
  return __builtin_amdgcn_wmma_f32_16x16x32_bf16(false, a, false, b,
                                                 (short)0, c, false, false);
}

// --------------------------- f32 -> bf16 copies -----------------------------
__global__ __launch_bounds__(256) void convert_k(const float* __restrict__ W,
                                                 const float* __restrict__ V,
                                                 __bf16* __restrict__ Wb,
                                                 __bf16* __restrict__ Vb) {
  int i = blockIdx.x * 256 + threadIdx.x;     // 2 * 512*512 threads total
  const int N = Hsz * Isz;
  if (i < N) {
    Wb[i] = to_bf16(W[i]);
  } else if (i < 2 * N) {
    Vb[i - N] = to_bf16(V[i - N]);
  }
}

// --------------------------- Phase 1: Wx GEMM -------------------------------
// One block (8 waves) per 16-row M tile. x tile is loaded + converted to bf16
// in LDS once; each wave then computes 4 N tiles of 16 cols, K=512.
#define XSTR 520                               // padded LDS row stride (bf16)

__global__ __launch_bounds__(256) void wx_gemm_k(const float* __restrict__ x,
                                                 const __bf16* __restrict__ Wb,
                                                 float* __restrict__ Wx) {
  __shared__ __bf16 xs[16 * XSTR];

  const int tid  = threadIdx.x;
  const int lane = tid & 31;
  const int wav  = tid >> 5;
  const int m0   = blockIdx.x * 16;            // 4000 M tiles
  const int n0   = wav * 64;                   // 4 N tiles per wave

  // cooperative stage: 16 rows x 512 f32 -> bf16 in LDS (float4 granularity)
  for (int i = tid; i < 16 * 128; i += 256) {
    const int r  = i >> 7;                     // row 0..15
    const int c4 = i & 127;                    // float4 index 0..127
    float4 f = *(const float4*)(x + (size_t)(m0 + r) * Isz + c4 * 4);
    unsigned long long pk =
        (unsigned long long)bf16_bits(f.x)
      | ((unsigned long long)bf16_bits(f.y) << 16)
      | ((unsigned long long)bf16_bits(f.z) << 32)
      | ((unsigned long long)bf16_bits(f.w) << 48);
    *(unsigned long long*)&xs[r * XSTR + c4 * 4] = pk;
  }
  __syncthreads();

  const int khalf = lane >> 4;
  const int colB  = lane & 15;
  const int rowA  = lane & 15;

  v8f acc0 = zero8(), acc1 = zero8(), acc2 = zero8(), acc3 = zero8();

  for (int k0 = 0; k0 < Isz; k0 += 32) {
    AFrag a;
    a.h[0] = *(const v8bf*)&xs[rowA * XSTR + k0 + khalf * 16];
    a.h[1] = *(const v8bf*)&xs[rowA * XSTR + k0 + khalf * 16 + 8];

    const int kb = k0 + khalf * 16;
    v16bf b0 = *(const v16bf*)(Wb + (size_t)(n0 +      colB) * Isz + kb);
    v16bf b1 = *(const v16bf*)(Wb + (size_t)(n0 + 16 + colB) * Isz + kb);
    v16bf b2 = *(const v16bf*)(Wb + (size_t)(n0 + 32 + colB) * Isz + kb);
    v16bf b3 = *(const v16bf*)(Wb + (size_t)(n0 + 48 + colB) * Isz + kb);

    acc0 = wmma_bf16(a.v, b0, acc0);
    acc1 = wmma_bf16(a.v, b1, acc1);
    acc2 = wmma_bf16(a.v, b2, acc2);
    acc3 = wmma_bf16(a.v, b3, acc3);
  }

#pragma unroll
  for (int r = 0; r < 8; ++r) {
    const int m = m0 + r + khalf * 8;
    float* orow = Wx + (size_t)m * Hsz + n0 + colB;
    orow[0]  = acc0[r];
    orow[16] = acc1[r];
    orow[32] = acc2[r];
    orow[48] = acc3[r];
  }
}

// --------------------------- Phase 2: recurrent scan ------------------------
// Grid = 2 blocks x 512 threads (16 waves). Block bi owns batch rows
// [16*bi, 16*bi+16). Wave w owns output columns [32*w, 32*w+32) (2 N tiles).
// o (prev spikes, bf16) lives in double-buffered LDS; v lives in registers
// in C/D layout; V (bf16) streams from L2 (0.5 MB, fully resident).
#define OSTR 520                               // padded LDS row stride (bf16)
#define OBUF (16 * OSTR)

__global__ __launch_bounds__(512) void scan_k(const float* __restrict__ Wx,
                                              const float* __restrict__ noise,
                                              const float* __restrict__ vthr,
                                              const __bf16* __restrict__ Vb,
                                              float* __restrict__ out) {
  __shared__ __bf16 obuf[2 * OBUF];

  const int lane  = threadIdx.x & 31;
  const int wav   = threadIdx.x >> 5;
  const int khalf = lane >> 4;
  const int colB  = lane & 15;
  const int rowA  = lane & 15;                 // A-operand row within tile
  const int n0    = wav * 32;                  // this wave's first column
  const int b0    = blockIdx.x * 16;           // first batch row of block

  // zero both o buffers (o_0 = 0)
  for (int i = threadIdx.x; i < 2 * OBUF; i += 512)
    obuf[i] = to_bf16(0.0f);
  __syncthreads();

  const float thr0 = vthr[n0 + colB];
  const float thr1 = vthr[n0 + 16 + colB];

  v8f v0 = zero8(), v1 = zero8();              // membrane state (C/D layout)
  int p = 0;

  for (int t = 0; t < Tsz; ++t) {
    // prefetch next step's Wx / noise rows into cache while WMMAs run
    if (t + 1 < Tsz) {
      __builtin_prefetch(Wx    + ((size_t)(b0 + rowA) * Tsz + (t + 1)) * Hsz + n0 + colB, 0, 1);
      __builtin_prefetch(noise + ((size_t)(t + 1) * Bsz + b0 + rowA)   * Hsz + n0 + colB, 0, 1);
    }

    // i_fast = o @ V.T for this wave's 2 N tiles
    v8f acc0 = zero8(), acc1 = zero8();
    const __bf16* ob = &obuf[p * OBUF];
    for (int k0 = 0; k0 < Hsz; k0 += 32) {
      AFrag a;
      a.h[0] = *(const v8bf*)(ob + rowA * OSTR + k0 + khalf * 16);
      a.h[1] = *(const v8bf*)(ob + rowA * OSTR + k0 + khalf * 16 + 8);

      const int kb = k0 + khalf * 16;
      v16bf bv0 = *(const v16bf*)(Vb + (size_t)(n0 +      colB) * Hsz + kb);
      v16bf bv1 = *(const v16bf*)(Vb + (size_t)(n0 + 16 + colB) * Hsz + kb);
      acc0 = wmma_bf16(a.v, bv0, acc0);
      acc1 = wmma_bf16(a.v, bv1, acc1);
    }

    // elementwise Euler update + spike, write o to out and to LDS (other buf)
    __bf16* onext = &obuf[(1 - p) * OBUF];
#pragma unroll
    for (int r = 0; r < 8; ++r) {
      const int mrow = r + khalf * 8;          // 0..15 within batch tile
      const int b    = b0 + mrow;
      const size_t rowBT = (size_t)b * Tsz + t;            // [B,T,*] row
      const size_t rowTB = (size_t)t * Bsz + b;            // [T,B,*] row

      const float wx0 = Wx[rowBT * Hsz + n0 + colB];
      const float wx1 = Wx[rowBT * Hsz + n0 + 16 + colB];
      const float nz0 = noise[rowTB * Hsz + n0 + colB];
      const float nz1 = noise[rowTB * Hsz + n0 + 16 + colB];

      v0[r] = DECAY * v0[r] + HSTEP * (wx0 + acc0[r]) + NSCALE * nz0;
      v1[r] = DECAY * v1[r] + HSTEP * (wx1 + acc1[r]) + NSCALE * nz1;

      const float o0 = (v0[r] > thr0) ? INVH : 0.0f;
      const float o1 = (v1[r] > thr1) ? INVH : 0.0f;

      out[rowBT * Hsz + n0 + colB]      = o0;
      out[rowBT * Hsz + n0 + 16 + colB] = o1;
      onext[mrow * OSTR + n0 + colB]      = to_bf16(o0);
      onext[mrow * OSTR + n0 + 16 + colB] = to_bf16(o1);
    }
    __syncthreads();
    p ^= 1;
  }
}

// ---------------------------------------------------------------------------
extern "C" void kernel_launch(void* const* d_in, const int* in_sizes, int n_in,
                              void* d_out, int out_size, void* d_ws, size_t ws_size,
                              hipStream_t stream) {
  (void)in_sizes; (void)n_in; (void)out_size; (void)ws_size;

  const float* x     = (const float*)d_in[0];   // [B,T,I]
  const float* W     = (const float*)d_in[1];   // [H,I]
  const float* V     = (const float*)d_in[2];   // [H,H]
  const float* vthr  = (const float*)d_in[3];   // [H]
  const float* noise = (const float*)d_in[4];   // [T,B,H]
  float*       out   = (float*)d_out;           // [B,T,H]

  // workspace layout
  char* ws = (char*)d_ws;
  float*  Wx = (float*)ws;                                    // 64000*512*4 B
  __bf16* Wb = (__bf16*)(ws + (size_t)Bsz * Tsz * Hsz * 4);   // 512*512*2 B
  __bf16* Vb = Wb + (size_t)Hsz * Isz;                        // 512*512*2 B

  // 1) f32 -> bf16 copies of W and V
  convert_k<<<(2 * Hsz * Isz + 255) / 256, 256, 0, stream>>>(W, V, Wb, Vb);

  // 2) Wx = x @ W.T  (4000 blocks; x tile staged in LDS, 4 N-tiles per wave)
  wx_gemm_k<<<4000, 256, 0, stream>>>(x, Wb, Wx);

  // 3) recurrent scan: 2 blocks of 16 waves
  scan_k<<<2, 512, 0, stream>>>(Wx, noise, vthr, Vb, out);
}
